// MambaFull_85418309583392
// MI455X (gfx1250) — matrix operand order repaired and checked by reading, not compiled, not run on hardware
//
#include <hip/hip_runtime.h>
#include <hip/hip_bf16.h>
#include <stdint.h>

#define DEV __device__ __forceinline__

// ---- model dims (match reference) ----
constexpr int D_MODEL = 1024;
constexpr int D_INNER = 2048;
constexpr int D_XZ    = 2 * D_INNER;            // 4096
constexpr int D_STATE = 16;
constexpr int DT_RANK = 64;
constexpr int D_DBL   = DT_RANK + 2 * D_STATE;  // 96
constexpr int N_LAYER = 4;
constexpr int VOCAB   = 32000;
constexpr int BB      = 2;
constexpr int LL      = 1024;
constexpr int T_TOK   = BB * LL;                // 2048 tokens

// ---- CDNA5 WMMA types ----
typedef __attribute__((ext_vector_type(16))) __bf16 v16bf;
typedef __attribute__((ext_vector_type(8)))  __bf16 v8bf;
typedef __attribute__((ext_vector_type(8)))  float  v8f;

DEV __bf16 f32_to_bf16(float f) {
  unsigned int u = __builtin_bit_cast(unsigned int, f);
  u += 0x7fffu + ((u >> 16) & 1u);           // round-to-nearest-even
  unsigned short h = (unsigned short)(u >> 16);
  return __builtin_bit_cast(__bf16, h);
}

DEV v16bf cat8(v8bf lo, v8bf hi) {
  return __builtin_shufflevector(lo, hi, 0,1,2,3,4,5,6,7,8,9,10,11,12,13,14,15);
}

DEV float softplus_f(float x) { return (x > 20.f) ? x : log1pf(__expf(x)); }
DEV float silu_f(float x)     { return x / (1.f + __expf(-x)); }
DEV int   imin(int a, int b)  { return a < b ? a : b; }

// ---- CDNA5 async global->LDS copy (ASYNCcnt-tracked, ISA ch10/§15.18) ----
// Generic LDS pointer low 32 bits are the wave-relative LDS byte offset.
DEV void async_to_lds_b128(void* lds, const void* g) {
  unsigned loff = (unsigned)(uintptr_t)lds;
  asm volatile("global_load_async_to_lds_b128 %0, %1, off"
               :: "v"(loff), "v"(g) : "memory");
}
DEV void wait_async0() { asm volatile("s_wait_asynccnt 0" ::: "memory"); }

// =====================================================================
// bf16 WMMA GEMM: C[M,N] = A[M,K] * Bt[N,K]^T  (both operands already
// bf16; Bt is the weight pre-transposed to N-major so both A and B LDS
// tiles stage as contiguous 16B chunks via GLOBAL_LOAD_ASYNC_TO_LDS_B128,
// double-buffered against the WMMA consume of the previous K-tile).
// Block tile 64(M) x 128(N), 4 waves (2x2), wave tile 32x64 = 2x4 WMMA.
// OOB rows/cols are *clamped* (not zero-filled): they only ever feed
// output elements that the guarded epilogue never stores, so staging is
// branch-free straight-line async issue (no exec-mask churn).
// op: 0 = none, 1 = softplus(x + bias[col]), 2 = x + resid[row,col]
// Optional Cbf: also emit a bf16 copy of the output (feeds next GEMM).
// =====================================================================
__global__ __launch_bounds__(128)
void gemm_bf16_wmma(const __bf16* __restrict__ A, int lda,
                    const __bf16* __restrict__ Bt,
                    float* C, __bf16* Cbf, int M, int N, int K,
                    const float* __restrict__ bias,
                    const float* resid, int op) {
  __shared__ __align__(16) __bf16 As[2][64][40];   // 64 rows x 32 k (+pad)
  __shared__ __align__(16) __bf16 Bs[2][128][40];  // 128 cols x 32 k (+pad)

  const int tid   = threadIdx.x;
  const int lane  = tid & 31;
  const int wave  = tid >> 5;
  const int wm    = wave >> 1;        // wave row 0..1 (32 rows each)
  const int wn    = wave & 1;         // wave col 0..1 (64 cols each)
  const int m16   = lane & 15;
  const int khalf = lane >> 4;        // half-wave select
  const int bM    = blockIdx.y * 64;
  const int bN    = blockIdx.x * 128;

  v8f acc[2][4] = {};

  const int NK = K >> 5;              // K-tiles of 32

  // per-thread staging coordinates (clamped once, outside the loop)
  const int ar  = (tid * 2) >> 2;                 // A row for chunk pair
  const int acc0 = ((tid * 2) & 3) * 8;           // A col of first chunk
  const size_t aRow = (size_t)imin(bM + ar, M - 1) * lda;
  const int bn  = tid;                            // B col (4 chunks per col)
  const size_t bRow = (size_t)imin(bN + bn, N - 1) * K;

  auto stage = [&](int sel, int k0) {
    // A tile: 64 rows x 64B -> 256 chunks, 2 consecutive per thread
    async_to_lds_b128((void*)&As[sel][ar][acc0],     A + aRow + k0 + acc0);
    async_to_lds_b128((void*)&As[sel][ar][acc0 + 8], A + aRow + k0 + acc0 + 8);
    // B tile: 128 cols x 64B -> 512 chunks, one full col per thread
#pragma unroll
    for (int q = 0; q < 4; ++q)
      async_to_lds_b128((void*)&Bs[sel][bn][q * 8], Bt + bRow + k0 + q * 8);
  };

  stage(0, 0);
  wait_async0();
  __syncthreads();

  for (int kt = 0; kt < NK; ++kt) {
    const int cur = kt & 1;
    if (kt + 1 < NK) {
      stage(cur ^ 1, (kt + 1) * 32);  // overlap copy of next tile
      __builtin_prefetch(A + aRow + (kt + 2) * 32, 0, 3);
      __builtin_prefetch(Bt + bRow + (kt + 2) * 32, 0, 3);
    }

    // ---- fragments per ISA 7.12.2 16-bit layouts ----
    // A 16x32: lane (m16,khalf) holds K = khalf*8+{0..7}, 16+khalf*8+{0..7}
    v16bf afrag[2];
#pragma unroll
    for (int i = 0; i < 2; ++i) {
      const int r = wm * 32 + i * 16 + m16;
      afrag[i] = cat8(*(const v8bf*)&As[cur][r][khalf * 8],
                      *(const v8bf*)&As[cur][r][16 + khalf * 8]);
    }
    // B 32x16: lane (n16,khalf) holds K = khalf*8+{0..7}, 16+khalf*8+{0..7} of column n16
    v16bf bfrag[4];
#pragma unroll
    for (int j = 0; j < 4; ++j) {
      const int n = wn * 64 + j * 16 + m16;
      bfrag[j] = cat8(*(const v8bf*)&Bs[cur][n][khalf * 8],
                      *(const v8bf*)&Bs[cur][n][16 + khalf * 8]);
    }
#pragma unroll
    for (int i = 0; i < 2; ++i)
#pragma unroll
      for (int j = 0; j < 4; ++j)
        acc[i][j] = __builtin_amdgcn_wmma_f32_16x16x32_bf16(
            false, afrag[i], false, bfrag[j], (short)0, acc[i][j],
            false, false);

    wait_async0();
    __syncthreads();
  }

  // ---- epilogue: C/D layout — VGPR v holds row khalf*8+v, lane n16 = col
#pragma unroll
  for (int i = 0; i < 2; ++i)
#pragma unroll
    for (int j = 0; j < 4; ++j) {
      const int col = bN + wn * 64 + j * 16 + m16;
      if (col < N) {
#pragma unroll
        for (int v = 0; v < 8; ++v) {
          const int row = bM + wm * 32 + i * 16 + khalf * 8 + v;
          if (row < M) {
            float val = acc[i][j][v];
            if (op == 1)      val = softplus_f(val + bias[col]);
            else if (op == 2) val += resid[(size_t)row * N + col];
            C[(size_t)row * N + col] = val;
            if (Cbf) Cbf[(size_t)row * N + col] = f32_to_bf16(val);
          }
        }
      }
    }
}

// =====================================================================
// Weight convert+transpose: wt[n][k] = bf16(w[k][n]).  32x32 LDS tile,
// coalesced on both sides.  K,N multiples of 32 for all model weights.
// =====================================================================
__global__ __launch_bounds__(256)
void wt_convert_kernel(const float* __restrict__ w, __bf16* __restrict__ wt,
                       int K, int N) {
  __shared__ float tile[32][33];
  const int k0 = blockIdx.x * 32;
  const int n0 = blockIdx.y * 32;
  const int c  = threadIdx.x & 31;
  const int r0 = (threadIdx.x >> 5) * 4;
#pragma unroll
  for (int i = 0; i < 4; ++i)
    tile[r0 + i][c] = w[(size_t)(k0 + r0 + i) * N + n0 + c];
  __syncthreads();
#pragma unroll
  for (int i = 0; i < 4; ++i)
    wt[(size_t)(n0 + r0 + i) * K + k0 + c] = f32_to_bf16(tile[c][r0 + i]);
}

// =====================================================================
// Embedding gather: one block per token, 256 thr x float4
// =====================================================================
__global__ __launch_bounds__(256)
void embed_kernel(const int* __restrict__ tokens,
                  const float* __restrict__ emb, float* __restrict__ out) {
  const int row = blockIdx.x;
  const int tok = tokens[row];
  const float4* src = (const float4*)(emb + (size_t)tok * D_MODEL);
  float4* dst = (float4*)(out + (size_t)row * D_MODEL);
  dst[threadIdx.x] = src[threadIdx.x];
}

// =====================================================================
// RMSNorm: one block (256 thr) per row; emits bf16 (GEMM operand)
// =====================================================================
__global__ __launch_bounds__(256)
void rmsnorm_kernel(const float* __restrict__ x, const float* __restrict__ w,
                    __bf16* __restrict__ out, int D) {
  const int row = blockIdx.x;
  const float* xr = x + (size_t)row * D;
  float s = 0.f;
  for (int i = threadIdx.x; i < D; i += 256) { float v = xr[i]; s += v * v; }
#pragma unroll
  for (int off = 16; off > 0; off >>= 1) s += __shfl_down(s, off, 32);
  __shared__ float red[8];
  if ((threadIdx.x & 31) == 0) red[threadIdx.x >> 5] = s;
  __syncthreads();
  if (threadIdx.x == 0) {
    float t = 0.f;
#pragma unroll
    for (int i = 0; i < 8; ++i) t += red[i];
    red[0] = rsqrtf(t / (float)D + 1e-5f);
  }
  __syncthreads();
  const float scale = red[0];
  for (int i = threadIdx.x; i < D; i += 256)
    out[(size_t)row * D + i] = f32_to_bf16(xr[i] * scale * w[i]);
}

// =====================================================================
// Causal depthwise conv (D_CONV=4) + SiLU. u lives in xz[:, 0:D_INNER].
// Emits f32 (scan input) and bf16 (x_proj GEMM operand).
// =====================================================================
__global__ __launch_bounds__(256)
void conv_silu_kernel(const float* __restrict__ xz,
                      const float* __restrict__ w,
                      const float* __restrict__ bias,
                      float* __restrict__ out,
                      __bf16* __restrict__ out_bf) {
  const int id = blockIdx.x * 256 + threadIdx.x;
  const int d  = id % D_INNER;
  const int t  = (id / D_INNER) % LL;
  const int b  = id / (D_INNER * LL);
  float acc = bias[d];
#pragma unroll
  for (int j = 0; j < 4; ++j) {
    const int tt = t - 3 + j;
    if (tt >= 0)
      acc += xz[((size_t)(b * LL + tt)) * D_XZ + d] * w[d * 4 + j];
  }
  const float r = silu_f(acc);
  const size_t o = ((size_t)(b * LL + t)) * D_INNER + d;
  out[o]    = r;
  out_bf[o] = f32_to_bf16(r);
}

// =====================================================================
// Selective scan: one thread per (b, d); 16 SSM states in registers,
// sequential over L. Bm/Cm fetched as float4 (b128 loads, broadcast via
// L2 across the 2048 channels of a batch). Folds in skip (u*D) and
// silu(z) gate; emits bf16 for the out_proj GEMM.
// =====================================================================
__global__ __launch_bounds__(256)
void scan_kernel(const float* __restrict__ u, const float* __restrict__ dt,
                 const float* __restrict__ A_log, const float* __restrict__ dbl,
                 const float* __restrict__ Dp, const float* __restrict__ xz,
                 __bf16* __restrict__ yout) {
  const int id = blockIdx.x * 256 + threadIdx.x;   // B*Di threads
  const int d  = id % D_INNER;
  const int b  = id / D_INNER;
  float a[D_STATE], h[D_STATE];
#pragma unroll
  for (int n = 0; n < D_STATE; ++n) {
    a[n] = -__expf(A_log[(size_t)d * D_STATE + n]);
    h[n] = 0.f;
  }
  const float Dd = Dp[d];
  const size_t base = (size_t)b * LL;
  for (int t = 0; t < LL; ++t) {
    const size_t r  = base + t;
    const float dtv = dt[r * D_INNER + d];
    const float uv  = u[r * D_INNER + d];
    float4 B4[4], C4[4];
    {
      const float4* Bp = (const float4*)(dbl + r * D_DBL + DT_RANK);
      const float4* Cp = (const float4*)(dbl + r * D_DBL + DT_RANK + D_STATE);
#pragma unroll
      for (int q = 0; q < 4; ++q) { B4[q] = Bp[q]; C4[q] = Cp[q]; }
    }
    const float* Bv = (const float*)B4;
    const float* Cv = (const float*)C4;
    float y = 0.f;
#pragma unroll
    for (int n = 0; n < D_STATE; ++n) {
      const float dA = __expf(dtv * a[n]);
      h[n] = dA * h[n] + dtv * Bv[n] * uv;
      y   += h[n] * Cv[n];
    }
    const float z = xz[r * D_XZ + D_INNER + d];
    yout[r * D_INNER + d] = f32_to_bf16((y + uv * Dd) * silu_f(z));
  }
}

// =====================================================================
static void launch_gemm(const __bf16* A, int lda, const __bf16* Bt,
                        float* C, __bf16* Cbf, int M, int N, int K,
                        const float* bias, const float* resid, int op,
                        hipStream_t s) {
  dim3 grid((N + 127) / 128, (M + 63) / 64);
  gemm_bf16_wmma<<<grid, dim3(128), 0, s>>>(A, lda, Bt, C, Cbf, M, N, K,
                                            bias, resid, op);
}

static void launch_wt(const float* w, __bf16* wt, int K, int N, hipStream_t s) {
  wt_convert_kernel<<<dim3(K / 32, N / 32), 256, 0, s>>>(w, wt, K, N);
}

extern "C" void kernel_launch(void* const* d_in, const int* in_sizes, int n_in,
                              void* d_out, int out_size, void* d_ws, size_t ws_size,
                              hipStream_t stream) {
  (void)in_sizes; (void)n_in; (void)out_size; (void)ws_size;
  const int*   tokens   = (const int*)  d_in[0];
  const float* emb      = (const float*)d_in[1];
  const float* norm_w   = (const float*)d_in[2];
  const float* in_proj  = (const float*)d_in[3];
  const float* conv_w   = (const float*)d_in[4];
  const float* conv_b   = (const float*)d_in[5];
  const float* x_proj   = (const float*)d_in[6];
  const float* dt_w     = (const float*)d_in[7];
  const float* dt_b     = (const float*)d_in[8];
  const float* A_log    = (const float*)d_in[9];
  const float* Dp       = (const float*)d_in[10];
  const float* out_proj = (const float*)d_in[11];
  const float* fnorm_w  = (const float*)d_in[12];
  const float* head_w   = (const float*)d_in[13];
  float* out = (float*)d_out;

  // ---- workspace layout (~163 MB) ----
  char* p = (char*)d_ws;
  auto alloc = [&](size_t bytes) {
    void* q = p; p += (bytes + 255) & ~(size_t)255; return q;
  };
  float*  x        = (float*) alloc((size_t)T_TOK * D_MODEL * 4);
  float*  xz       = (float*) alloc((size_t)T_TOK * D_XZ    * 4);
  float*  uconv    = (float*) alloc((size_t)T_TOK * D_INNER * 4);
  float*  dbl      = (float*) alloc((size_t)T_TOK * D_DBL   * 4);
  float*  dtb      = (float*) alloc((size_t)T_TOK * D_INNER * 4);
  __bf16* xn_bf    = (__bf16*)alloc((size_t)T_TOK * D_MODEL * 2);
  __bf16* uconv_bf = (__bf16*)alloc((size_t)T_TOK * D_INNER * 2);
  __bf16* dbl_bf   = (__bf16*)alloc((size_t)T_TOK * D_DBL   * 2);
  __bf16* yg_bf    = (__bf16*)alloc((size_t)T_TOK * D_INNER * 2);
  __bf16* wt       = (__bf16*)alloc((size_t)VOCAB * D_MODEL * 2); // max weight

  embed_kernel<<<T_TOK, 256, 0, stream>>>(tokens, emb, x);

  for (int i = 0; i < N_LAYER; ++i) {
    rmsnorm_kernel<<<T_TOK, 256, 0, stream>>>(x, norm_w + (size_t)i * D_MODEL,
                                              xn_bf, D_MODEL);
    // xz = xn @ in_proj[i]        [2048,1024]x[1024,4096]
    launch_wt(in_proj + (size_t)i * D_MODEL * D_XZ, wt, D_MODEL, D_XZ, stream);
    launch_gemm(xn_bf, D_MODEL, wt, xz, nullptr,
                T_TOK, D_XZ, D_MODEL, nullptr, nullptr, 0, stream);
    // u = silu(conv1d(xz[:, :Di]))
    conv_silu_kernel<<<(T_TOK * D_INNER) / 256, 256, 0, stream>>>(
        xz, conv_w + (size_t)i * D_INNER * 4, conv_b + (size_t)i * D_INNER,
        uconv, uconv_bf);
    // dbl = u @ x_proj[i]         [2048,2048]x[2048,96]
    launch_wt(x_proj + (size_t)i * D_INNER * D_DBL, wt, D_INNER, D_DBL, stream);
    launch_gemm(uconv_bf, D_INNER, wt, dbl, dbl_bf,
                T_TOK, D_DBL, D_INNER, nullptr, nullptr, 0, stream);
    // dt = softplus(dbl[:, :64] @ dt_w[i] + dt_b[i])
    launch_wt(dt_w + (size_t)i * DT_RANK * D_INNER, wt, DT_RANK, D_INNER, stream);
    launch_gemm(dbl_bf, D_DBL, wt, dtb, nullptr,
                T_TOK, D_INNER, DT_RANK, dt_b + (size_t)i * D_INNER, nullptr, 1,
                stream);
    // selective scan + skip + silu(z) gate
    scan_kernel<<<(BB * D_INNER) / 256, 256, 0, stream>>>(
        uconv, dtb, A_log + (size_t)i * D_INNER * D_STATE, dbl,
        Dp + (size_t)i * D_INNER, xz, yg_bf);
    // x = yg @ out_proj[i] + x    [2048,2048]x[2048,1024]
    launch_wt(out_proj + (size_t)i * D_INNER * D_MODEL, wt, D_INNER, D_MODEL,
              stream);
    launch_gemm(yg_bf, D_INNER, wt, x, nullptr,
                T_TOK, D_MODEL, D_INNER, nullptr, x, 2, stream);
  }

  rmsnorm_kernel<<<T_TOK, 256, 0, stream>>>(x, fnorm_w, xn_bf, D_MODEL);
  // logits = xn @ head_w          [2048,1024]x[1024,32000]
  launch_wt(head_w, wt, D_MODEL, VOCAB, stream);
  launch_gemm(xn_bf, D_MODEL, wt, out, nullptr,
              T_TOK, VOCAB, D_MODEL, nullptr, nullptr, 0, stream);
}
